// DitPreprocess_53446573031486
// MI455X (gfx1250) — matrix operand order repaired and checked by reading, not compiled, not run on hardware
//
#include <hip/hip_runtime.h>

typedef __attribute__((ext_vector_type(16))) _Float16 v16h;
typedef __attribute__((ext_vector_type(8)))  float    v8f;

#define D_COND  512
#define D_CODE  512
#define D_SPK   192
#define D_MODEL 1024
#define T_CODE  2048
#define B_SZ    2
#define T_FULL  (2 * T_CODE)        // 4096
#define N_ROWS  (B_SZ * T_CODE)     // 4096 unique GEMM rows
#define K_DIM   D_CODE              // 512

#define OUT_EMB_ELEMS  (B_SZ * T_FULL * D_MODEL)      // 8,388,608
#define OUT_ROPE_ELEMS (2 * T_FULL * 64)              // 524,288
#define OUT_MASK_ELEMS (T_FULL * T_FULL)              // 16,777,216

// ---------------------------------------------------------------------------
// Kernel 0: per-batch bias  bias[b][d] = cond[b]·W_cond[:,d] + spk[b]·W_spk[:,d] + b[d]
// Tiny (2048 outputs), plain VALU.
// ---------------------------------------------------------------------------
__global__ void bias_kernel(const float* __restrict__ cond,
                            const float* __restrict__ spk,
                            const float* __restrict__ W,
                            const float* __restrict__ bvec,
                            float* __restrict__ bias) {
    int tid = blockIdx.x * blockDim.x + threadIdx.x;
    if (tid >= B_SZ * D_MODEL) return;
    int b = tid / D_MODEL;
    int d = tid - b * D_MODEL;
    const float* Wc = W;                                    // rows [0, 512)
    const float* Ws = W + (D_COND + D_CODE) * D_MODEL;      // rows [1024, 1216)
    const float* cb = cond + b * D_COND;
    const float* sb = spk + b * D_SPK;
    float acc = bvec[d];
    for (int c = 0; c < D_COND; ++c) acc += cb[c] * Wc[c * D_MODEL + d];
    for (int s = 0; s < D_SPK;  ++s) acc += sb[s] * Ws[s * D_MODEL + d];
    bias[tid] = acc;
}

// ---------------------------------------------------------------------------
// Kernel 1: WMMA GEMM. One wave32 per 16x16 output tile.
//   A = embed_table[code[r]] (gathered rows), B = W_code, f32->f16 on the fly,
//   accumulate f32 via v_wmma_f32_16x16x32_f16, K = 512 => 16 WMMA per tile.
//   K-loop unrolled x4 so loads of the next steps clause ahead of the WMMAs.
//   Each result row is written twice (the repeat(...,2) in the reference).
// ---------------------------------------------------------------------------
__global__ void __launch_bounds__(256)
gemm_wmma_kernel(const int* __restrict__ code,
                 const float* __restrict__ embed,
                 const float* __restrict__ W,
                 const float* __restrict__ bias,
                 float* __restrict__ out) {
    const float* Wcode = W + D_COND * D_MODEL;   // rows [512, 1024)

    const int lane   = threadIdx.x & 31;
    const int wave   = threadIdx.x >> 5;
    const int tile   = blockIdx.x * 8 + wave;    // 16384 tiles total
    const int tile_n = tile & 63;                // 64 tiles over D_MODEL
    const int tile_m = tile >> 6;                // 256 tiles over N_ROWS
    const int half   = lane >> 4;                // lane half (K split)
    const int mloc   = lane & 15;

    // A-side: this lane's embedding row (lanes 16-31 mirror rows of 0-15)
    const int    r    = tile_m * 16 + mloc;
    const float* arow = embed + (size_t)code[r] * D_CODE;

    const int col0 = tile_n * 16;
    const int krow = lane & 15;                  // B-side column N within 16-wide tile

    v8f acc = {};

    #pragma unroll 4
    for (int k0 = 0; k0 < K_DIM; k0 += 32) {
        // ---- A fragment (16x32 f16): lane half h holds K = k0+h*8+[0,8) and
        //      K = k0+16+h*8+[0,8), per ISA 16-bit A layout.
        const float4* pa0 = (const float4*)(arow + k0 + half * 8);
        const float4* pa1 = (const float4*)(arow + k0 + 16 + half * 8);
        float4 f0 = pa0[0], f1 = pa0[1];
        float4 f2 = pa1[0], f3 = pa1[1];

        // ---- B fragment (32x16 f16): lane holds column N = col0 + krow,
        //      K = k0+h*8+[0,8) and K = k0+16+h*8+[0,8) (same order as A).
        const float* brow = Wcode + (size_t)(k0 + half * 8) * D_MODEL + col0 + krow;
        float4 g0 = make_float4(brow[0 * D_MODEL], brow[1 * D_MODEL], brow[2 * D_MODEL], brow[3 * D_MODEL]);
        float4 g1 = make_float4(brow[4 * D_MODEL], brow[5 * D_MODEL], brow[6 * D_MODEL], brow[7 * D_MODEL]);
        float4 g2 = make_float4(brow[16 * D_MODEL], brow[17 * D_MODEL], brow[18 * D_MODEL], brow[19 * D_MODEL]);
        float4 g3 = make_float4(brow[20 * D_MODEL], brow[21 * D_MODEL], brow[22 * D_MODEL], brow[23 * D_MODEL]);

        // Prefetch next-K W rows (global_prefetch_b8). Always in-bounds:
        // max row index stays well inside W's 1216 rows.
        __builtin_prefetch(brow + 32 * D_MODEL, 0, 0);

        v16h a;
        a[0]  = (_Float16)f0.x; a[1]  = (_Float16)f0.y;
        a[2]  = (_Float16)f0.z; a[3]  = (_Float16)f0.w;
        a[4]  = (_Float16)f1.x; a[5]  = (_Float16)f1.y;
        a[6]  = (_Float16)f1.z; a[7]  = (_Float16)f1.w;
        a[8]  = (_Float16)f2.x; a[9]  = (_Float16)f2.y;
        a[10] = (_Float16)f2.z; a[11] = (_Float16)f2.w;
        a[12] = (_Float16)f3.x; a[13] = (_Float16)f3.y;
        a[14] = (_Float16)f3.z; a[15] = (_Float16)f3.w;

        v16h bfrag;
        bfrag[0]  = (_Float16)g0.x; bfrag[1]  = (_Float16)g0.y;
        bfrag[2]  = (_Float16)g0.z; bfrag[3]  = (_Float16)g0.w;
        bfrag[4]  = (_Float16)g1.x; bfrag[5]  = (_Float16)g1.y;
        bfrag[6]  = (_Float16)g1.z; bfrag[7]  = (_Float16)g1.w;
        bfrag[8]  = (_Float16)g2.x; bfrag[9]  = (_Float16)g2.y;
        bfrag[10] = (_Float16)g2.z; bfrag[11] = (_Float16)g2.w;
        bfrag[12] = (_Float16)g3.x; bfrag[13] = (_Float16)g3.y;
        bfrag[14] = (_Float16)g3.z; bfrag[15] = (_Float16)g3.w;

        acc = __builtin_amdgcn_wmma_f32_16x16x32_f16(
            /*neg_a=*/false, a, /*neg_b=*/false, bfrag,
            /*c_mod=*/(short)0, acc, /*reuse_a=*/false, /*reuse_b=*/false);
    }

    // ---- Writeout: C/D layout — VGPR i, lane l: M = i + 8*(l>>4), N = l&15.
    // Tiles never straddle the batch boundary (2048 % 16 == 0).
    const int btile = (tile_m * 16) / T_CODE;
    const int ncol  = col0 + mloc;
    const float bb  = bias[btile * D_MODEL + ncol];
    const int outb  = btile * (T_FULL * D_MODEL);
    #pragma unroll
    for (int i = 0; i < 8; ++i) {
        int row = tile_m * 16 + half * 8 + i;       // unique row
        int tc  = row - btile * T_CODE;
        float v = acc[i] + bb;
        int base = outb + (2 * tc) * D_MODEL + ncol;
        out[base]           = v;                     // t = 2*tc
        out[base + D_MODEL] = v;                     // t = 2*tc + 1
    }
}

// ---------------------------------------------------------------------------
// Kernel 2: rope[s][t][j] = (s ? sin : cos)(t * 10000^{-(2*(j>>1))/64})
// ---------------------------------------------------------------------------
__global__ void rope_kernel(float* __restrict__ out) {
    int gid = blockIdx.x * blockDim.x + threadIdx.x;
    if (gid >= OUT_ROPE_ELEMS) return;
    int j = gid & 63;
    int t = (gid >> 6) & (T_FULL - 1);
    int s = gid >> 18;                       // 64 * 4096 = 2^18 per s-plane
    float e = (float)(2 * (j >> 1)) * (1.0f / 64.0f);
    float theta = powf(10000.0f, -e);
    float ang = (float)t * theta;
    out[gid] = s ? sinf(ang) : cosf(ang);
}

// ---------------------------------------------------------------------------
// Kernel 3: mask[i][j] = j/24 - i/24 ; float4 stores, pure bandwidth.
// ---------------------------------------------------------------------------
__global__ void mask_kernel(float* __restrict__ out) {
    int gid = blockIdx.x * blockDim.x + threadIdx.x;   // 4,194,304 threads
    int i  = gid >> 10;                                // 1024 float4 per row
    int j0 = (gid & 1023) << 2;
    int bi = i / 24;
    float4 v;
    v.x = (float)((j0 + 0) / 24 - bi);
    v.y = (float)((j0 + 1) / 24 - bi);
    v.z = (float)((j0 + 2) / 24 - bi);
    v.w = (float)((j0 + 3) / 24 - bi);
    ((float4*)out)[gid] = v;
}

// ---------------------------------------------------------------------------
extern "C" void kernel_launch(void* const* d_in, const int* in_sizes, int n_in,
                              void* d_out, int out_size, void* d_ws, size_t ws_size,
                              hipStream_t stream) {
    const float* cond  = (const float*)d_in[0];
    const float* spk   = (const float*)d_in[1];
    const int*   code  = (const int*)d_in[2];
    const float* embed = (const float*)d_in[3];
    const float* W     = (const float*)d_in[4];
    const float* bvec  = (const float*)d_in[5];
    float* out  = (float*)d_out;
    float* bias = (float*)d_ws;                 // 2*1024 floats scratch

    bias_kernel<<<(B_SZ * D_MODEL + 255) / 256, 256, 0, stream>>>(cond, spk, W, bvec, bias);

    // 16384 tiles, 8 waves (tiles) per 256-thread block
    gemm_wmma_kernel<<<2048, 256, 0, stream>>>(code, embed, W, bias, out);

    rope_kernel<<<(OUT_ROPE_ELEMS + 255) / 256, 256, 0, stream>>>(out + OUT_EMB_ELEMS);

    mask_kernel<<<(OUT_MASK_ELEMS / 4 + 255) / 256, 256, 0, stream>>>(
        out + OUT_EMB_ELEMS + OUT_ROPE_ELEMS);
}
